// Ibp338V22Simplified_10179072491650
// MI455X (gfx1250) — matrix operand order repaired and checked by reading, not compiled, not run on hardware
//
#include <hip/hip_runtime.h>
#include <hip/hip_bf16.h>

typedef __attribute__((ext_vector_type(16))) _Float16 v16h;
typedef __attribute__((ext_vector_type(8)))  float    v8f;
typedef __attribute__((ext_vector_type(4)))  _Float16 v4h;
typedef __attribute__((ext_vector_type(4)))  float    v4f;

constexpr int TOK   = 17;   // real tokens
constexpr int MPAD  = 32;   // padded tokens (2 WMMA M-tiles)
constexpr int DMODEL= 48;   // model dim (3 N-tiles)
constexpr int DPAD  = 64;   // K padded to 2x32
constexpr int NH    = 3;    // heads
constexpr int DMLP  = 96;   // mlp hidden (6 N-tiles / 3 K-steps)
constexpr float ATT_SCALE = 0.25f;

struct Params {
    const float* p[43];
    float* out;
};

__device__ __forceinline__ v8f wmma16(v16h a, v16h b, v8f c) {
    return __builtin_amdgcn_wmma_f32_16x16x32_f16(
        /*neg_a=*/false, a, /*neg_b=*/false, b,
        /*c_mod=*/(short)0, c, /*reuse_a=*/false, /*reuse_b=*/false);
}

__device__ __forceinline__ v4h cvt4(v4f v) {
    v4h h;
    h[0] = (_Float16)v[0]; h[1] = (_Float16)v[1];
    h[2] = (_Float16)v[2]; h[3] = (_Float16)v[3];
    return h;
}

// A fragment: 16x32 f16 tile, rows r0..+15, cols k0..+31 (row stride ld).
__device__ __forceinline__ v16h ld_a(const _Float16* s, int ld, int r0, int k0) {
    int lane = threadIdx.x & 31;
    int r  = r0 + (lane & 15);
    int kb = k0 + ((lane & 16) ? 8 : 0);
    v16h a;
#pragma unroll
    for (int e = 0; e < 16; ++e) a[e] = s[r * ld + kb + (e < 8 ? e : e + 8)];
    return a;
}

// B fragment from row-major B[k][n] (lane%16 = n, elements over k in table order).
__device__ __forceinline__ v16h ld_b(const _Float16* s, int ld, int k0, int n0) {
    int lane = threadIdx.x & 31;
    int kb = k0 + ((lane & 16) ? 8 : 0);
    v16h b;
#pragma unroll
    for (int e = 0; e < 16; ++e) b[e] = s[(kb + (e < 8 ? e : e + 8)) * ld + n0 + (lane & 15)];
    return b;
}

// Stage global f32 W[rows x K] -> LDS f16 W^T dst[Kpad][rows], zero pad k>=K.
// Vectorized: one global_load_b128 (4 consecutive k) + 4 scattered ds_store_b16.
__device__ __forceinline__ void load_wT16(const float* g, int rows, int K, int Kpad,
                                          _Float16* dst, int tid) {
    int kg = Kpad >> 2;                       // k-groups of 4 per row
    for (int i = tid; i < rows * kg; i += 256) {
        int r = i / kg, k = (i - r * kg) << 2;
        v4f v = {0.0f, 0.0f, 0.0f, 0.0f};
        if (k < K) v = *(const v4f*)&g[r * K + k];
        v4h h = cvt4(v);
        dst[(k + 0) * rows + r] = h[0];
        dst[(k + 1) * rows + r] = h[1];
        dst[(k + 2) * rows + r] = h[2];
        dst[(k + 3) * rows + r] = h[3];
    }
}

// f32 LDS -> f16 LDS, 4 elems/iter (ds_load_b128 + ds_store_b64)
__device__ __forceinline__ void cvt_x16(const float* sx, _Float16* xh, int tid) {
    for (int gq = tid; gq < (MPAD * DPAD) / 4; gq += 256)
        *(v4h*)&xh[gq * 4] = cvt4(*(const v4f*)&sx[gq * 4]);
}

// Y[32 x Nout] (ldy) (+)= A[32 x K] @ W^T + bias ; Wt in LDS as [Kpad x Nout] f16
__device__ __forceinline__ void gemm_f32out(const _Float16* A, int lda,
                                            const _Float16* Wt, int ldn,
                                            const float* bias, float* Y, int ldy,
                                            int ntiles, int ksteps, bool accum, int tid) {
    int wave = tid >> 5, lane = tid & 31;
    int jobs = 2 * ntiles;
    for (int t = wave; t < jobs; t += 8) {
        int mt = t / ntiles, nt = t - mt * ntiles;
        v8f c = {};
        for (int ks = 0; ks < ksteps; ++ks)
            c = wmma16(ld_a(A, lda, mt * 16, ks * 32), ld_b(Wt, ldn, ks * 32, nt * 16), c);
        int col = nt * 16 + (lane & 15);
        int r0  = mt * 16 + ((lane & 16) ? 8 : 0);
        float bb = bias ? bias[col] : 0.0f;
#pragma unroll
        for (int v = 0; v < 8; ++v) {
            float val = c[v] + bb;
            float* yp = &Y[(r0 + v) * ldy + col];
            if (accum) *yp += val; else *yp = val;
        }
    }
}

__global__ __launch_bounds__(256)
void vit_fused_kernel(Params P) {
    const int tid  = threadIdx.x;
    const int wave = tid >> 5;
    const int lane = tid & 31;

    __shared__ __align__(16) float    s_x [MPAD * DPAD];    // activations f32
    __shared__ __align__(16) float    sc32[MPAD * DMLP];    // input img / scores / mlp h
    __shared__ __align__(16) _Float16 xh  [MPAD * DPAD];    // f16 x (A operand)
    __shared__ __align__(16) _Float16 wldsT[DPAD * DMLP];   // W^T staging (2x 64x48, or 64x96)
    __shared__ __align__(16) _Float16 qph [NH * MPAD * 32]; // per-head Q [tok][d32]
    __shared__ __align__(16) _Float16 kphT[NH * MPAD * 32]; // per-head K^T [d32][tok]
    __shared__ __align__(16) _Float16 vh  [MPAD * DPAD];    // V, head h at cols 16h
    __shared__ __align__(16) _Float16 oh  [MPAD * DPAD];    // attention context f16
    __shared__ __align__(16) _Float16 sc16[MPAD * DMLP];    // attn probs / relu(h) f16

    const float* inp = P.p[0];
    const float* cw  = P.p[1];
    const float* cb  = P.p[2];
    const float* cls = P.p[3];
    const float* pos = P.p[4];

    // ---- async DMA: input image (3*32*32 f32 = 12KB) -> sc32 via
    // global_load_async_to_lds_b128 (ASYNCcnt); overlaps the zero-init below.
    for (int i = tid; i < 768; i += 256) {               // 768 x 16B
        const float* gsrc = inp + i * 4;
        unsigned int  ldst = (unsigned int)(unsigned long long)(&sc32[i * 4]);
        asm volatile("global_load_async_to_lds_b128 %0, %1, off"
                     :: "v"(ldst), "v"(gsrc) : "memory");
    }

    // ---- zero-init padded regions, 128-bit stores (overlaps async DMA) ----
    {
        v4h zh = {(_Float16)0.0f, (_Float16)0.0f, (_Float16)0.0f, (_Float16)0.0f};
        v4f zf = {0.0f, 0.0f, 0.0f, 0.0f};
        for (int g = tid; g < (MPAD * DPAD) / 4; g += 256) {
            *(v4f*)&s_x[g * 4] = zf;
            *(v4h*)&xh[g * 4] = zh; *(v4h*)&vh[g * 4] = zh; *(v4h*)&oh[g * 4] = zh;
        }
        for (int g = tid; g < (NH * MPAD * 32) / 4; g += 256) {
            *(v4h*)&qph[g * 4] = zh; *(v4h*)&kphT[g * 4] = zh; *(v4h*)&sc16[g * 4] = zh;
        }
    }
    asm volatile("s_wait_asynccnt 0x0" ::: "memory");
    __syncthreads();

    // ---- patch embed: 16x192x48 GEMM in 3 K-chunks of 64 (chunk == channel) ----
    v8f pacc = {};
    for (int ch = 0; ch < 3; ++ch) {
        // A chunk [16][64]: 8 contiguous floats per (patch, kh) from LDS image
        for (int g = tid; g < 128; g += 256) {
            int pt = g >> 3, kh = g & 7;
            int pr = pt >> 2, pc = pt & 3;
            const float* src = &sc32[ch * 1024 + (pr * 8 + kh) * 32 + pc * 8];
            *(v4h*)&xh[pt * DPAD + kh * 8]     = cvt4(*(const v4f*)src);
            *(v4h*)&xh[pt * DPAD + kh * 8 + 4] = cvt4(*(const v4f*)(src + 4));
        }
        // W^T chunk [64][48]: global_load_b128 + 4 scattered b16 stores
        for (int g = tid; g < 48 * 16; g += 256) {
            int o = g >> 4, kk = (g & 15) << 2;
            v4h h = cvt4(*(const v4f*)&cw[o * 192 + ch * 64 + kk]);
            wldsT[(kk + 0) * DMODEL + o] = h[0];
            wldsT[(kk + 1) * DMODEL + o] = h[1];
            wldsT[(kk + 2) * DMODEL + o] = h[2];
            wldsT[(kk + 3) * DMODEL + o] = h[3];
        }
        __syncthreads();
        if (wave < 3) {
            for (int ks = 0; ks < 2; ++ks)
                pacc = wmma16(ld_a(xh, DPAD, 0, ks * 32),
                              ld_b(wldsT, DMODEL, ks * 32, wave * 16), pacc);
        }
        __syncthreads();
    }
    if (wave < 3) {  // patches -> token rows 1..16, + conv bias + pos emb
        int col = wave * 16 + (lane & 15);
        int r0  = (lane & 16) ? 8 : 0;
#pragma unroll
        for (int v = 0; v < 8; ++v) {
            int row = r0 + v + 1;
            s_x[row * DPAD + col] = pacc[v] + cb[col] + pos[row * DMODEL + col];
        }
    }
    if (tid < DMODEL) s_x[tid] = cls[tid] + pos[tid];    // cls token row 0
    __syncthreads();

    // ---- 3 transformer blocks ----
    for (int blk = 0; blk < 3; ++blk) {
        const float* const* bp = &P.p[5 + blk * 12];
        // bp: 0 wv 1 bv 2 wk 3 bk 4 wq 5 bq 6 wo 7 bo 8 w1 9 b1 10 w2 11 b2

        // prefetch next block's weights (global_prefetch_b8, no counters)
        if (blk < 2) {
            const float* const* np = &P.p[5 + (blk + 1) * 12];
            const int widx  [6] = {0, 2, 4, 6, 8, 10};
            const int wbytes[6] = {9216, 9216, 9216, 9216, 18432, 18432};
#pragma unroll
            for (int w = 0; w < 6; ++w)
                if (tid * 256 < wbytes[w])
                    __builtin_prefetch((const char*)np[widx[w]] + tid * 256, 0, 0);
        }

        cvt_x16(s_x, xh, tid);
        // stage V^T (offset 0) and K^T (offset 3072) together
        load_wT16(bp[0], DMODEL, DMODEL, DPAD, wldsT, tid);
        load_wT16(bp[2], DMODEL, DMODEL, DPAD, wldsT + 3072, tid);
        __syncthreads();

        // V + K projections fused: 12 tile jobs across 8 waves
        for (int t = wave; t < 12; t += 8) {
            bool isV = t < 6;
            int tt = isV ? t : t - 6;
            int mt = tt / 3, nt = tt - mt * 3;
            const _Float16* Wt  = isV ? wldsT : wldsT + 3072;
            const float*    bia = isV ? bp[1] : bp[3];
            v8f c = {};
            for (int ks = 0; ks < 2; ++ks)
                c = wmma16(ld_a(xh, DPAD, mt * 16, ks * 32),
                           ld_b(Wt, DMODEL, ks * 32, nt * 16), c);
            int nl = lane & 15;
            int r0 = mt * 16 + ((lane & 16) ? 8 : 0);
            float bb = bia[nt * 16 + nl];
#pragma unroll
            for (int v = 0; v < 8; ++v) {
                float val = c[v] + bb;
                if (isV) vh  [(r0 + v) * DPAD + nt * 16 + nl] = (_Float16)val; // [tok][col]
                else     kphT[nt * 1024 + nl * 32 + (r0 + v)] = (_Float16)val; // [d][tok]
            }
        }
        __syncthreads();

        // stage Q^T (offset 0) and wo^T (offset 3072, consumed after attention)
        load_wT16(bp[4], DMODEL, DMODEL, DPAD, wldsT, tid);
        load_wT16(bp[6], DMODEL, DMODEL, DPAD, wldsT + 3072, tid);
        __syncthreads();

        // Q projection: 6 tile jobs, per-head padded layout [tok][d32]
        for (int t = wave; t < 6; t += 8) {
            int mt = t / 3, nt = t - mt * 3;
            v8f c = {};
            for (int ks = 0; ks < 2; ++ks)
                c = wmma16(ld_a(xh, DPAD, mt * 16, ks * 32),
                           ld_b(wldsT, DMODEL, ks * 32, nt * 16), c);
            int nl = lane & 15;
            int r0 = mt * 16 + ((lane & 16) ? 8 : 0);
            float bb = bp[5][nt * 16 + nl];
#pragma unroll
            for (int v = 0; v < 8; ++v)
                qph[nt * 1024 + (r0 + v) * 32 + nl] = (_Float16)(c[v] + bb);
        }
        __syncthreads();

        // scores[h] = SCALE * Q_h @ K_h^T (K=16 zero-padded to 32): 12 tile jobs
        for (int j = wave; j < 12; j += 8) {
            int h = j >> 2, t = j & 3, mt = t >> 1, nt = t & 1;
            v8f c = {};
            c = wmma16(ld_a(qph + h * 1024, 32, mt * 16, 0),
                       ld_b(kphT + h * 1024, 32, 0, nt * 16), c);
            int col = nt * 16 + (lane & 15);
            int r0  = mt * 16 + ((lane & 16) ? 8 : 0);
#pragma unroll
            for (int v = 0; v < 8; ++v)
                sc32[h * 1024 + (r0 + v) * 32 + col] = c[v] * ATT_SCALE;
        }
        __syncthreads();

        // softmax over n=0..16, masked; full f16 prob rows (pad -> 0)
        if (tid < NH * MPAD) {
            int h = tid >> 5, m = tid & 31;
            const float* sp = &sc32[h * 1024 + m * 32];
            _Float16*    ap = &sc16[h * 1024 + m * 32];
            if (m < TOK) {
                float mx = -1e30f;
                for (int n = 0; n < TOK; ++n) mx = fmaxf(mx, sp[n]);
                float sum = 0.0f;
                for (int n = 0; n < TOK; ++n) sum += __expf(sp[n] - mx);
                float inv = 1.0f / sum;
                for (int n = 0; n < TOK; ++n) ap[n] = (_Float16)(__expf(sp[n] - mx) * inv);
                for (int n = TOK; n < 32; ++n) ap[n] = (_Float16)0.0f;
            } else {
                for (int n = 0; n < 32; ++n) ap[n] = (_Float16)0.0f;
            }
        }
        __syncthreads();

        // context = attn @ V (K = tokens, zero-masked past 17): 6 tile jobs
        for (int j = wave; j < 6; j += 8) {
            int h = j >> 1, mt = j & 1;
            v8f c = {};
            c = wmma16(ld_a(sc16 + h * 1024, 32, mt * 16, 0),
                       ld_b(vh, DPAD, 0, h * 16), c);
            int col = h * 16 + (lane & 15);
            int r0  = mt * 16 + ((lane & 16) ? 8 : 0);
#pragma unroll
            for (int v = 0; v < 8; ++v)
                oh[(r0 + v) * DPAD + col] = (_Float16)c[v];
        }
        __syncthreads();

        // attn output projection (+residual into s_x); wo^T already staged
        gemm_f32out(oh, DPAD, wldsT + 3072, DMODEL, bp[7], s_x, DPAD, 3, 2, true, tid);
        __syncthreads();

        // MLP
        cvt_x16(s_x, xh, tid);
        load_wT16(bp[8], DMLP, DMODEL, DPAD, wldsT, tid);            // w1^T [64][96]
        __syncthreads();
        gemm_f32out(xh, DPAD, wldsT, DMLP, bp[9], sc32, DMLP, 6, 2, false, tid);
        __syncthreads();
        for (int g = tid; g < (MPAD * DMLP) / 4; g += 256) {         // relu + cvt, 128-bit
            v4f v = *(const v4f*)&sc32[g * 4];
            v[0] = fmaxf(v[0], 0.0f); v[1] = fmaxf(v[1], 0.0f);
            v[2] = fmaxf(v[2], 0.0f); v[3] = fmaxf(v[3], 0.0f);
            *(v4h*)&sc16[g * 4] = cvt4(v);
        }
        __syncthreads();
        load_wT16(bp[10], DMODEL, DMLP, DMLP, wldsT, tid);           // w2^T [96][48]
        __syncthreads();
        gemm_f32out(sc16, DMLP, wldsT, DMODEL, bp[11], s_x, DPAD, 3, 3, true, tid);
        __syncthreads();
    }

    // ---- mean pool over 17 tokens + head ----
    if (tid < DMODEL) {
        float s = 0.0f;
        for (int m = 0; m < TOK; ++m) s += s_x[m * DPAD + tid];
        sc32[tid] = s * (1.0f / 17.0f);
    }
    __syncthreads();
    if (tid < 10) {
        const float* hw = P.p[41];
        const float* hb = P.p[42];
        float s = hb[tid];
        for (int d = 0; d < DMODEL; ++d) s += sc32[d] * hw[tid * DMODEL + d];
        P.out[tid] = s;
    }
}

extern "C" void kernel_launch(void* const* d_in, const int* in_sizes, int n_in,
                              void* d_out, int out_size, void* d_ws, size_t ws_size,
                              hipStream_t stream) {
    (void)in_sizes; (void)out_size; (void)d_ws; (void)ws_size;
    Params P;
    for (int i = 0; i < 43; ++i) P.p[i] = (i < n_in) ? (const float*)d_in[i] : nullptr;
    P.out = (float*)d_out;
    hipLaunchKernelGGL(vit_fused_kernel, dim3(1), dim3(256), 0, stream, P);
}
